// LowPassCTRNN_26972394619444
// MI455X (gfx1250) — matrix operand order repaired and checked by reading, not compiled, not run on hardware
//
#include <hip/hip_runtime.h>

typedef __attribute__((ext_vector_type(16))) __bf16 v16bf;
typedef __attribute__((ext_vector_type(8)))  float  v8f;

namespace {

constexpr int BATCH  = 64;
constexpr int TSTEPS = 1000;
constexpr int DIN    = 128;
constexpr int DREC   = 512;
constexpr int DOUT   = 64;
constexpr float ALPHA = 0.1f;   // DT / TAU

union Frag {
  unsigned u[8];
  uint4    q[2];
  v16bf    h;
};

__device__ __forceinline__ unsigned pack_bf16(float x, float y) {
  unsigned ux = __float_as_uint(x);
  unsigned uy = __float_as_uint(y);
  ux = (ux + 0x7FFFu + ((ux >> 16) & 1u)) >> 16;   // RNE
  uy = (uy + 0x7FFFu + ((uy >> 16) & 1u)) >> 16;
  return ux | (uy << 16);
}

__device__ __forceinline__ unsigned bf16_1(float x) {
  unsigned u = __float_as_uint(x);
  return (u + 0x7FFFu + ((u >> 16) & 1u)) >> 16;
}

// Dword index of element (row m, col k) inside a pre-packed 16-bit A/B
// fragment array (tiles of 256 dwords; ISA 16x32 16-bit layout:
// lanes0-15: K{0..7}=v0..3, K{16..23}=v4..7; lanes16-31: K{8..15}, K{24..31}).
__device__ __forceinline__ int frag_dword(int m, int k) {
  const int kt = k >> 5, kk = k & 31;
  return (kt << 8) + ((m + ((kk & 8) << 1)) << 3) + ((kk & 16) >> 2) + ((kk & 7) >> 1);
}

// Load one 256-dword fragment tile (works for LDS or global pointers).
__device__ __forceinline__ v16bf load_frag(const unsigned* __restrict__ fr,
                                           int tile, int lane) {
  const uint4* p = (const uint4*)(fr + (tile << 8) + (lane << 3));
  Frag f;
  f.q[0] = p[0];
  f.q[1] = p[1];
  return f.h;
}

__device__ __forceinline__ v8f wmma_bf16(v16bf a, v16bf b, v8f c) {
  return __builtin_amdgcn_wmma_f32_16x16x32_bf16(false, a, false, b,
                                                 (short)0, c, false, false);
}

__device__ __forceinline__ float retanh(float x) {
#if __has_builtin(__builtin_amdgcn_tanhf)
  return fmaxf(__builtin_amdgcn_tanhf(x), 0.0f);
#else
  return fmaxf(tanhf(x), 0.0f);
#endif
}

// Pack W [N rows x K cols, row-major f32] as B = W^T (KxN) into bf16 WMMA
// B-fragments; tile = nt*(K/32)+kt. ISA 32x16 16-bit B layout: lane&15 = n,
// lanes<16 hold K 0..7,16..23 (2/dword), lanes>=16 hold K 8..15,24..31.
__global__ void pack_b_kernel(const float* __restrict__ W,
                              unsigned* __restrict__ dst, int N, int K) {
  const int KT  = K >> 5;
  const int idx = blockIdx.x * blockDim.x + threadIdx.x;   // one packed dword
  if (idx >= (N * K) / 2) return;
  const int v    = idx & 7;
  const int lane = (idx >> 3) & 31;
  const int tile = idx >> 8;
  const int kt   = tile % KT;
  const int nt   = tile / KT;
  const int n = nt * 16 + (lane & 15);
  const int k = kt * 32 + ((lane >> 4) << 3) + ((v & 4) << 2) + ((v & 3) << 1);
  dst[idx] = pack_bf16(W[(size_t)n * K + k], W[(size_t)n * K + k + 1]);
}

__launch_bounds__(256, 1)
__global__ void ctrnn_main(const float* __restrict__ input,
                           const float* __restrict__ brn,
                           const float* __restrict__ bx,
                           const float* __restrict__ by,
                           const float* __restrict__ r0,
                           const unsigned* __restrict__ wrrB,
                           const unsigned* __restrict__ wrxB,
                           const unsigned* __restrict__ wyrB,
                           float* __restrict__ y_out,
                           float* __restrict__ r_out) {
  __shared__ unsigned r_frag[(DREC / 32) * 256];                 // 16 KB, r as bf16 A-frags
  __shared__ unsigned x_frag[(DIN / 32) * 256];                  //  4 KB, input A-frags
  __shared__ unsigned wyr_frag[(DREC / 32) * (DOUT / 16) * 256]; // 64 KB, Wyr B-frags

  const int tid  = threadIdx.x;
  const int lane = tid & 31;
  const int wv   = tid >> 5;           // wave wv owns n-tiles wv*4 .. wv*4+3
  const int b0   = blockIdx.x * 16;    // 16-batch tile

  // One-time: Wyr fragments to LDS.
  for (int i = tid; i < (DREC / 32) * (DOUT / 16) * 64; i += 256)
    ((uint4*)wyr_frag)[i] = ((const uint4*)wyrB)[i];

  // One-time: initial r fragments from r0 (broadcast over batch rows).
  for (int idx = tid; idx < 16 * (DREC / 8); idx += 256) {
    const int m = idx >> 6, kb = (idx & 63) << 3;
    float4 a = *(const float4*)(r0 + kb);
    float4 b = *(const float4*)(r0 + kb + 4);
    uint4 d;
    d.x = pack_bf16(a.x, a.y);  d.y = pack_bf16(a.z, a.w);
    d.z = pack_bf16(b.x, b.y);  d.w = pack_bf16(b.z, b.w);
    *(uint4*)(r_frag + frag_dword(m, kb)) = d;
  }

  // f32 recurrent state lives in registers at this thread's D-fragment slots.
  v8f rreg[4];
#pragma unroll
  for (int j = 0; j < 4; ++j) {
    const float rv = r0[(wv * 4 + j) * 16 + (lane & 15)];
#pragma unroll
    for (int v = 0; v < 8; ++v) rreg[j][v] = rv;
  }

  for (int t = 0; t < TSTEPS; ++t) {
    // ---- stage input tile [16 x 128] straight into bf16 A-fragments -------
    {
      const int m = tid >> 4, kb = (tid & 15) << 3;   // one 8-col chunk/thread
      const float* src = input + ((size_t)(b0 + m) * TSTEPS + t) * DIN + kb;
      float4 a = *(const float4*)src;
      float4 b = *(const float4*)(src + 4);
      uint4 d;
      d.x = pack_bf16(a.x, a.y);  d.y = pack_bf16(a.z, a.w);
      d.z = pack_bf16(b.x, b.y);  d.w = pack_bf16(b.z, b.w);
      *(uint4*)(x_frag + frag_dword(m, kb)) = d;
    }
    __syncthreads();   // BAR1: x_frag ready; r_frag writes (init/prev ph3) visible

    // ---- phase 1: acc = input @ Wrx^T + bx --------------------------------
    v8f acc[4];
#pragma unroll
    for (int j = 0; j < 4; ++j) {
      const float bxv = bx[(wv * 4 + j) * 16 + (lane & 15)];
#pragma unroll
      for (int v = 0; v < 8; ++v) acc[j][v] = bxv;
    }
    // Bounded unroll: keep only ~2 iterations of fragments in flight so the
    // scheduler overlaps L2 latency without spilling hoisted loads to scratch.
#pragma unroll 2
    for (int kt = 0; kt < DIN / 32; ++kt) {
      const v16bf aF = load_frag(x_frag, kt, lane);
#pragma unroll
      for (int j = 0; j < 4; ++j) {
        const v16bf bF = load_frag(wrxB, (wv * 4 + j) * (DIN / 32) + kt, lane);
        acc[j] = wmma_bf16(aF, bF, acc[j]);
      }
    }

    // ---- phase 2: acc += r @ Wrr^T  (stream Wrr B-frags from L2) ----------
#pragma unroll 2
    for (int kt = 0; kt < DREC / 32; ++kt) {
      const v16bf aF = load_frag(r_frag, kt, lane);
      if (kt + 1 < DREC / 32)
        __builtin_prefetch(wrrB + (((wv * 4) * 16 + kt + 1) << 8) + (lane << 3), 0, 1);
#pragma unroll
      for (int j = 0; j < 4; ++j) {
        const v16bf bF = load_frag(wrrB, (wv * 4 + j) * 16 + kt, lane);
        acc[j] = wmma_bf16(aF, bF, acc[j]);
      }
    }
    __syncthreads();   // BAR2: all r_frag reads complete before state update

    // ---- phase 3: retanh + leaky update; r -> regs, d_out, bf16 frags -----
    unsigned short* rb = (unsigned short*)r_frag;
#pragma unroll
    for (int j = 0; j < 4; ++j) {
      const int col = (wv * 4 + j) * 16 + (lane & 15);
      const int mh  = (lane >> 4) << 3;
      // half-index of element (mh, col) in r_frag; +16 halves per row
      const int kk    = col & 31;
      const int hbase = (frag_dword(mh, col) << 1) + (kk & 1);
#pragma unroll
      for (int v = 0; v < 8; ++v) {
        const int m = v + mh;
        const float f = retanh(acc[j][v]);
        const size_t gi = ((size_t)(b0 + m) * TSTEPS + t) * DREC + col;
        const float brv  = brn[gi];
        const float rold = rreg[j][v];
        const float rn   = rold + ALPHA * (f + brv - rold);
        rreg[j][v] = rn;
        r_out[gi]  = rn;
        rb[hbase + v * 16] = (unsigned short)bf16_1(rn);
      }
    }
    __syncthreads();   // BAR3: new r fragments visible

    // ---- phase 4: y = r_new @ Wyr^T + by  (waves 0..3, 16 cols each) ------
    if (wv < 4) {
      v8f yacc;
      const float byv = by[wv * 16 + (lane & 15)];
#pragma unroll
      for (int v = 0; v < 8; ++v) yacc[v] = byv;
#pragma unroll 4
      for (int kt = 0; kt < DREC / 32; ++kt) {
        const v16bf aF = load_frag(r_frag, kt, lane);
        const v16bf bF = load_frag(wyr_frag, wv * 16 + kt, lane);
        yacc = wmma_bf16(aF, bF, yacc);
      }
      const int col = wv * 16 + (lane & 15);
      const int mh  = (lane >> 4) << 3;
#pragma unroll
      for (int v = 0; v < 8; ++v) {
        const int m = v + mh;
        y_out[((size_t)(b0 + m) * TSTEPS + t) * DOUT + col] = yacc[v];
      }
    }
  }
}

} // anonymous namespace

extern "C" void kernel_launch(void* const* d_in, const int* in_sizes, int n_in,
                              void* d_out, int out_size, void* d_ws, size_t ws_size,
                              hipStream_t stream) {
  (void)in_sizes; (void)n_in; (void)out_size; (void)ws_size;
  const float* input = (const float*)d_in[0];
  const float* brn   = (const float*)d_in[1];
  const float* Wrx   = (const float*)d_in[2];
  const float* bx    = (const float*)d_in[3];
  const float* Wrr   = (const float*)d_in[4];
  const float* Wyr   = (const float*)d_in[5];
  const float* by    = (const float*)d_in[6];
  const float* r0    = (const float*)d_in[7];

  // Workspace: pre-packed bf16 WMMA B-fragments (Wrr 512KB, Wrx 128KB, Wyr 64KB)
  unsigned* wrrB = (unsigned*)d_ws;
  unsigned* wrxB = wrrB + (DREC * DREC / 2);
  unsigned* wyrB = wrxB + (DREC * DIN / 2);

  float* y_out = (float*)d_out;
  float* r_out = y_out + (size_t)BATCH * TSTEPS * DOUT;

  pack_b_kernel<<<(DREC * DREC / 2 + 255) / 256, 256, 0, stream>>>(Wrr, wrrB, DREC, DREC);
  pack_b_kernel<<<(DREC * DIN  / 2 + 255) / 256, 256, 0, stream>>>(Wrx, wrxB, DREC, DIN);
  pack_b_kernel<<<(DOUT * DREC / 2 + 255) / 256, 256, 0, stream>>>(Wyr, wyrB, DOUT, DREC);

  ctrnn_main<<<BATCH / 16, 256, 0, stream>>>(input, brn, bx, by, r0,
                                             wrrB, wrxB, wyrB, y_out, r_out);
}
